// MultiHeadAttention_80238579024572
// MI455X (gfx1250) — matrix operand order repaired and checked by reading, not compiled, not run on hardware
//
#include <hip/hip_runtime.h>
#include <hip/hip_bf16.h>
#include <math.h>

typedef __attribute__((ext_vector_type(16))) __bf16       v16bf;
typedef __attribute__((ext_vector_type(8)))  float        v8f;
typedef __attribute__((ext_vector_type(8)))  unsigned int u32x8;

#define D_MODEL 1024
#define HEADS   16
#define DKK     64
#define SEQ     2048
#define BATCH   4
#define ROWS    (BATCH * SEQ) /* 8192 */

__device__ __forceinline__ unsigned short f2bf(float f) {
  unsigned int u = __float_as_uint(f);
  u += 0x7FFFu + ((u >> 16) & 1u);            // round-to-nearest-even
  return (unsigned short)(u >> 16);
}

__device__ __forceinline__ v8f vzero8() {
  v8f v = {0.f, 0.f, 0.f, 0.f, 0.f, 0.f, 0.f, 0.f};
  return v;
}

// Load 16 bf16 (two 16B chunks) into a WMMA fragment (global or LDS source).
__device__ __forceinline__ v16bf load_frag16(const unsigned short* p0,
                                             const unsigned short* p1) {
  uint4 a = *reinterpret_cast<const uint4*>(p0);
  uint4 b = *reinterpret_cast<const uint4*>(p1);
  u32x8 r = {a.x, a.y, a.z, a.w, b.x, b.y, b.z, b.w};
  return __builtin_bit_cast(v16bf, r);
}

__device__ __forceinline__ v16bf ones_frag() {
  u32x8 r = {0x3F803F80u, 0x3F803F80u, 0x3F803F80u, 0x3F803F80u,
             0x3F803F80u, 0x3F803F80u, 0x3F803F80u, 0x3F803F80u};
  return __builtin_bit_cast(v16bf, r);   // 16x bf16(1.0)
}

__device__ __forceinline__ v8f wmma_bf16(v16bf a, v16bf b, v8f c) {
  return __builtin_amdgcn_wmma_f32_16x16x32_bf16(
      /*neg_a=*/false, a, /*neg_b=*/false, b,
      /*c_mod=*/(short)0, c, /*reuse_a=*/false, /*reuse_b=*/false);
}

// Async DMA: 16 bytes global -> LDS, tracked by ASYNCcnt (CDNA5).
__device__ __forceinline__ void async_b128(const unsigned short* gptr,
                                           void* lptr) {
  unsigned laddr = (unsigned)(uintptr_t)lptr;
  asm volatile("global_load_async_to_lds_b128 %0, %1, off"
               :: "v"(laddr), "v"(gptr)
               : "memory");
}

#define WAIT_ASYNCCNT(n) asm volatile("s_wait_asynccnt " #n ::: "memory")

// ---------------------------------------------------------------- convert
__global__ __launch_bounds__(256)
void cvt_bf16(const float* __restrict__ in, unsigned short* __restrict__ out,
              int n4) {
  int i = blockIdx.x * 256 + threadIdx.x;
  if (i < n4) {
    float4 f = reinterpret_cast<const float4*>(in)[i];
    ushort4 o;
    o.x = f2bf(f.x); o.y = f2bf(f.y); o.z = f2bf(f.z); o.w = f2bf(f.w);
    reinterpret_cast<ushort4*>(out)[i] = o;
  }
}

// ---------------------------------------------------------- QKV projection
// 4-wave workgroup computes 64(M) x 128(N); X tile async-staged in LDS
// (double buffered), each wave handles a 32-wide N slice (2 B tiles).
__global__ __launch_bounds__(128)
void proj_qkv(const unsigned short* __restrict__ Xh,
              const unsigned short* __restrict__ Wqh,
              const unsigned short* __restrict__ Wkh,
              const unsigned short* __restrict__ Wvh,
              const float* __restrict__ bq,
              const float* __restrict__ bk,
              const float* __restrict__ bv,
              unsigned short* __restrict__ Qh,
              unsigned short* __restrict__ Kh,
              unsigned short* __restrict__ Vth) {
  __shared__ __align__(16) unsigned short xt[2][64][32];  // 8 KB

  const int tid  = threadIdx.x;
  const int lane = tid & 31;
  const int w    = tid >> 5;
  const int col  = lane & 15;
  const int hi   = (lane >> 4) & 1;
  const int m0   = blockIdx.x * 64;
  const int n0   = blockIdx.y * 128 + w * 32;
  const int z    = blockIdx.z;

  const unsigned short* W    = (z == 0) ? Wqh : (z == 1) ? Wkh : Wvh;
  const float*          bias = (z == 0) ? bq  : (z == 1) ? bk  : bv;

  v8f acc[2][4];
#pragma unroll
  for (int u = 0; u < 2; ++u)
#pragma unroll
    for (int t = 0; t < 4; ++t) acc[u][t] = vzero8();

  // prologue: stage first X tile (64 rows x 32 k), 2 chunks per lane
  {
#pragma unroll
    for (int i = 0; i < 2; ++i) {
      int c = tid + 128 * i;
      int row = c >> 2, ko = (c & 3) * 8;
      async_b128(Xh + (size_t)(m0 + row) * D_MODEL + ko, &xt[0][row][ko]);
    }
  }

  const unsigned short* wrow0 = W + (size_t)(n0 + col) * D_MODEL;
  const unsigned short* wrow1 = W + (size_t)(n0 + 16 + col) * D_MODEL;
  for (int k0 = 0; k0 < D_MODEL; k0 += 32) {
    const int buf = (k0 >> 5) & 1;
    if (k0 + 32 < D_MODEL) {
#pragma unroll
      for (int i = 0; i < 2; ++i) {
        int c = tid + 128 * i;
        int row = c >> 2, ko = (c & 3) * 8;
        async_b128(Xh + (size_t)(m0 + row) * D_MODEL + k0 + 32 + ko,
                   &xt[buf ^ 1][row][ko]);
      }
      WAIT_ASYNCCNT(2);   // oldest 2 (current buffer) done, next 2 in flight
    } else {
      WAIT_ASYNCCNT(0);
    }
    __syncthreads();

    v16bf bf0 = load_frag16(wrow0 + k0 + hi * 8, wrow0 + k0 + 16 + hi * 8);
    v16bf bf1 = load_frag16(wrow1 + k0 + hi * 8, wrow1 + k0 + 16 + hi * 8);
#pragma unroll
    for (int t = 0; t < 4; ++t) {
      v16bf afrag = load_frag16(&xt[buf][t * 16 + col][hi * 8],
                                &xt[buf][t * 16 + col][16 + hi * 8]);
      acc[0][t] = wmma_bf16(afrag, bf0, acc[0][t]);
      acc[1][t] = wmma_bf16(afrag, bf1, acc[1][t]);
    }
    __syncthreads();
  }

#pragma unroll
  for (int u = 0; u < 2; ++u) {
    const int   n   = n0 + u * 16 + col;
    const int   h   = n >> 6;
    const int   dk  = n & 63;
    const float bv_ = bias[n];
#pragma unroll
    for (int t = 0; t < 4; ++t) {
#pragma unroll
      for (int j = 0; j < 8; ++j) {
        int m = m0 + t * 16 + j + 8 * hi;  // global row over B*S
        int b = m >> 11;
        int s = m & (SEQ - 1);
        unsigned short val = f2bf(acc[u][t][j] + bv_);
        if (z == 2) {
          Vth[((size_t)(b * HEADS + h) * DKK + dk) * SEQ + s] = val;
        } else {
          unsigned short* dst = (z == 0) ? Qh : Kh;
          dst[((size_t)(b * HEADS + h) * SEQ + s) * DKK + dk] = val;
        }
      }
    }
  }
}

// ------------------------------------------------------------- attention
// 4-wave workgroup covers 128 queries of one (b,h): 2 q-tiles (32 rows)
// per wave. K/V tiles per 32-key step are async-staged (double buffered)
// and shared; row-sums come from an extra WMMA against an all-ones tile.
__global__ __launch_bounds__(128)
void attn(const unsigned short* __restrict__ Qh,
          const unsigned short* __restrict__ Kh,
          const unsigned short* __restrict__ Vth,
          unsigned short* __restrict__ Oh) {
  __shared__ __align__(16) unsigned short kt[2][32][64];     // 8 KB: [key][dk]
  __shared__ __align__(16) unsigned short vt[2][64][32];     // 8 KB: [dk][key]
  __shared__ __align__(16) unsigned short pt[4][2][16][32];  // 8 KB: P per wave/tile

  const int tid  = threadIdx.x;
  const int lane = tid & 31;
  const int w    = tid >> 5;
  const int col  = lane & 15;
  const int hi   = (lane >> 4) & 1;
  const int q0   = blockIdx.x * 128 + w * 32;
  const int h    = blockIdx.y;
  const int b    = blockIdx.z;

  const unsigned short* Qb = Qh + (size_t)(b * HEADS + h) * SEQ * DKK;
  const unsigned short* Kb = Kh + (size_t)(b * HEADS + h) * SEQ * DKK;
  const unsigned short* Vb = Vth + (size_t)(b * HEADS + h) * DKK * SEQ;

  // Q fragments: 2 q-tiles x (dk [0,32), [32,64))
  v16bf qf[2][2];
#pragma unroll
  for (int u = 0; u < 2; ++u) {
    const unsigned short* qrow = Qb + (size_t)(q0 + u * 16 + col) * DKK;
    qf[u][0] = load_frag16(qrow + 0 + hi * 8, qrow + 16 + hi * 8);
    qf[u][1] = load_frag16(qrow + 32 + hi * 8, qrow + 48 + hi * 8);
  }

  v8f acc[2][4];
#pragma unroll
  for (int u = 0; u < 2; ++u)
#pragma unroll
    for (int d = 0; d < 4; ++d) acc[u][d] = vzero8();
  float mrow[2][8], lrow[2][8];
#pragma unroll
  for (int u = 0; u < 2; ++u)
#pragma unroll
    for (int j = 0; j < 8; ++j) { mrow[u][j] = -INFINITY; lrow[u][j] = 0.f; }

  // prologue: stage K/V for first 32-key step (4 async b128 per lane)
  {
#pragma unroll
    for (int i = 0; i < 2; ++i) {
      int c = tid + 128 * i;
      int row = c >> 3, ko = (c & 7) * 8;
      async_b128(Kb + (size_t)row * DKK + ko, &kt[0][row][ko]);
    }
#pragma unroll
    for (int i = 0; i < 2; ++i) {
      int c = tid + 128 * i;
      int dk = c >> 2, to = (c & 3) * 8;
      async_b128(Vb + (size_t)dk * SEQ + to, &vt[0][dk][to]);
    }
  }

  for (int t0 = 0; t0 < SEQ; t0 += 32) {
    const int buf = (t0 >> 5) & 1;
    if (t0 + 32 < SEQ) {
#pragma unroll
      for (int i = 0; i < 2; ++i) {
        int c = tid + 128 * i;
        int row = c >> 3, ko = (c & 7) * 8;
        async_b128(Kb + (size_t)(t0 + 32 + row) * DKK + ko,
                   &kt[buf ^ 1][row][ko]);
      }
#pragma unroll
      for (int i = 0; i < 2; ++i) {
        int c = tid + 128 * i;
        int dk = c >> 2, to = (c & 3) * 8;
        async_b128(Vb + (size_t)dk * SEQ + t0 + 32 + to, &vt[buf ^ 1][dk][to]);
      }
      WAIT_ASYNCCNT(4);   // oldest 4 (current buffers) done, next 4 in flight
    } else {
      WAIT_ASYNCCNT(0);
    }
    __syncthreads();

    // ---- scores = Q K^T for keys [t0, t0+32), K fragments reused by both q-tiles
    v8f s[2][2];
#pragma unroll
    for (int u = 0; u < 2; ++u) { s[u][0] = vzero8(); s[u][1] = vzero8(); }
#pragma unroll
    for (int kk = 0; kk < 2; ++kk) {
      v16bf kf0 = load_frag16(&kt[buf][col][kk * 32 + hi * 8],
                              &kt[buf][col][kk * 32 + 16 + hi * 8]);
      v16bf kf1 = load_frag16(&kt[buf][16 + col][kk * 32 + hi * 8],
                              &kt[buf][16 + col][kk * 32 + 16 + hi * 8]);
#pragma unroll
      for (int u = 0; u < 2; ++u) {
        s[u][0] = wmma_bf16(qf[u][kk], kf0, s[u][0]);
        s[u][1] = wmma_bf16(qf[u][kk], kf1, s[u][1]);
      }
    }

    // ---- online softmax per q-tile (row max via shfl; row sum via WMMA below)
    float corr[2][8];
#pragma unroll
    for (int u = 0; u < 2; ++u) {
#pragma unroll
      for (int j = 0; j < 8; ++j) {
        float a0 = s[u][0][j] * 0.125f;  // 1/sqrt(64)
        float a1 = s[u][1][j] * 0.125f;
        float mx = fmaxf(a0, a1);
#pragma unroll
        for (int msk = 1; msk < 16; msk <<= 1)
          mx = fmaxf(mx, __shfl_xor(mx, msk, 32));
        float mnew = fmaxf(mrow[u][j], mx);
        corr[u][j] = __expf(mrow[u][j] - mnew);
        mrow[u][j] = mnew;
        float p0 = __expf(a0 - mnew);
        float p1 = __expf(a1 - mnew);
#pragma unroll
        for (int d = 0; d < 4; ++d) acc[u][d][j] *= corr[u][j];
        pt[w][u][j + 8 * hi][col]      = f2bf(p0);
        pt[w][u][j + 8 * hi][16 + col] = f2bf(p1);
      }
    }
    // P regions are private per wave; in-wave DS ordering guarantees RAW.

    // ---- row sums (P @ ones) and O += P @ V, V fragments shared by q-tiles
    v16bf pf[2];
#pragma unroll
    for (int u = 0; u < 2; ++u) {
      pf[u] = load_frag16(&pt[w][u][col][hi * 8], &pt[w][u][col][16 + hi * 8]);
      v8f rsum = wmma_bf16(pf[u], ones_frag(), vzero8());
#pragma unroll
      for (int j = 0; j < 8; ++j)
        lrow[u][j] = lrow[u][j] * corr[u][j] + rsum[j];
    }
#pragma unroll
    for (int d = 0; d < 4; ++d) {
      v16bf vf = load_frag16(&vt[buf][d * 16 + col][hi * 8],
                             &vt[buf][d * 16 + col][16 + hi * 8]);
#pragma unroll
      for (int u = 0; u < 2; ++u) acc[u][d] = wmma_bf16(pf[u], vf, acc[u][d]);
    }
    __syncthreads();  // all waves done reading buf before it is refilled
  }

  // ---- normalize and store O as [B,S,D_MODEL] bf16
#pragma unroll
  for (int u = 0; u < 2; ++u) {
#pragma unroll
    for (int j = 0; j < 8; ++j) {
      float inv = 1.0f / lrow[u][j];
      int srow = q0 + u * 16 + j + 8 * hi;
#pragma unroll
      for (int d = 0; d < 4; ++d) {
        Oh[((size_t)(b * SEQ + srow)) * D_MODEL + h * DKK + d * 16 + col] =
            f2bf(acc[u][d][j] * inv);
      }
    }
  }
}

// -------------------------------------------------------- output projection
__global__ __launch_bounds__(128)
void proj_out(const unsigned short* __restrict__ Ohh,
              const unsigned short* __restrict__ Woh,
              const float* __restrict__ bo,
              float* __restrict__ out) {
  __shared__ __align__(16) unsigned short xt[2][64][32];  // 8 KB

  const int tid  = threadIdx.x;
  const int lane = tid & 31;
  const int w    = tid >> 5;
  const int col  = lane & 15;
  const int hi   = (lane >> 4) & 1;
  const int m0   = blockIdx.x * 64;
  const int n0   = blockIdx.y * 128 + w * 32;

  v8f acc[2][4];
#pragma unroll
  for (int u = 0; u < 2; ++u)
#pragma unroll
    for (int t = 0; t < 4; ++t) acc[u][t] = vzero8();

  {
#pragma unroll
    for (int i = 0; i < 2; ++i) {
      int c = tid + 128 * i;
      int row = c >> 2, ko = (c & 3) * 8;
      async_b128(Ohh + (size_t)(m0 + row) * D_MODEL + ko, &xt[0][row][ko]);
    }
  }

  const unsigned short* wrow0 = Woh + (size_t)(n0 + col) * D_MODEL;
  const unsigned short* wrow1 = Woh + (size_t)(n0 + 16 + col) * D_MODEL;
  for (int k0 = 0; k0 < D_MODEL; k0 += 32) {
    const int buf = (k0 >> 5) & 1;
    if (k0 + 32 < D_MODEL) {
#pragma unroll
      for (int i = 0; i < 2; ++i) {
        int c = tid + 128 * i;
        int row = c >> 2, ko = (c & 3) * 8;
        async_b128(Ohh + (size_t)(m0 + row) * D_MODEL + k0 + 32 + ko,
                   &xt[buf ^ 1][row][ko]);
      }
      WAIT_ASYNCCNT(2);
    } else {
      WAIT_ASYNCCNT(0);
    }
    __syncthreads();

    v16bf bf0 = load_frag16(wrow0 + k0 + hi * 8, wrow0 + k0 + 16 + hi * 8);
    v16bf bf1 = load_frag16(wrow1 + k0 + hi * 8, wrow1 + k0 + 16 + hi * 8);
#pragma unroll
    for (int t = 0; t < 4; ++t) {
      v16bf afrag = load_frag16(&xt[buf][t * 16 + col][hi * 8],
                                &xt[buf][t * 16 + col][16 + hi * 8]);
      acc[0][t] = wmma_bf16(afrag, bf0, acc[0][t]);
      acc[1][t] = wmma_bf16(afrag, bf1, acc[1][t]);
    }
    __syncthreads();
  }

#pragma unroll
  for (int u = 0; u < 2; ++u) {
    const int   n   = n0 + u * 16 + col;
    const float bv_ = bo[n];
#pragma unroll
    for (int t = 0; t < 4; ++t) {
#pragma unroll
      for (int j = 0; j < 8; ++j) {
        int m = m0 + t * 16 + j + 8 * hi;
        out[(size_t)m * D_MODEL + n] = acc[u][t][j] + bv_;
      }
    }
  }
}

// ------------------------------------------------------------------ launch
extern "C" void kernel_launch(void* const* d_in, const int* in_sizes, int n_in,
                              void* d_out, int out_size, void* d_ws,
                              size_t ws_size, hipStream_t stream) {
  const float* X  = (const float*)d_in[0];
  const float* Wq = (const float*)d_in[1];
  const float* bq = (const float*)d_in[2];
  const float* Wk = (const float*)d_in[3];
  const float* bk = (const float*)d_in[4];
  const float* Wv = (const float*)d_in[5];
  const float* bv = (const float*)d_in[6];
  const float* Wo = (const float*)d_in[7];
  const float* bo = (const float*)d_in[8];
  float* out = (float*)d_out;

  char* ws = (char*)d_ws;
  const size_t XH_BYTES = (size_t)ROWS * D_MODEL * 2;     // 16.78 MB
  const size_t WH_BYTES = (size_t)D_MODEL * D_MODEL * 2;  // 2.10 MB
  size_t off = 0;
  unsigned short* Xh  = (unsigned short*)(ws + off); off += XH_BYTES;
  unsigned short* Wqh = (unsigned short*)(ws + off); off += WH_BYTES;
  unsigned short* Wkh = (unsigned short*)(ws + off); off += WH_BYTES;
  unsigned short* Wvh = (unsigned short*)(ws + off); off += WH_BYTES;
  unsigned short* Woh = (unsigned short*)(ws + off); off += WH_BYTES;
  unsigned short* Qh  = (unsigned short*)(ws + off); off += XH_BYTES;
  unsigned short* Kh  = (unsigned short*)(ws + off); off += XH_BYTES;
  unsigned short* Vth = (unsigned short*)(ws + off); off += XH_BYTES;
  unsigned short* Oh  = (unsigned short*)(ws + off); off += XH_BYTES;

  // 1) convert inputs to bf16 (vectorized x4)
  {
    int n4 = ROWS * D_MODEL / 4;
    cvt_bf16<<<(n4 + 255) / 256, 256, 0, stream>>>(X, Xh, n4);
    int nw4 = D_MODEL * D_MODEL / 4;
    cvt_bf16<<<(nw4 + 255) / 256, 256, 0, stream>>>(Wq, Wqh, nw4);
    cvt_bf16<<<(nw4 + 255) / 256, 256, 0, stream>>>(Wk, Wkh, nw4);
    cvt_bf16<<<(nw4 + 255) / 256, 256, 0, stream>>>(Wv, Wvh, nw4);
    cvt_bf16<<<(nw4 + 255) / 256, 256, 0, stream>>>(Wo, Woh, nw4);
  }

  // 2) fused QKV projections (z = 0/1/2 -> Q/K/V)
  {
    dim3 g(ROWS / 64, D_MODEL / 128, 3);
    proj_qkv<<<g, 128, 0, stream>>>(Xh, Wqh, Wkh, Wvh, bq, bk, bv, Qh, Kh, Vth);
  }

  // 3) flash attention (4 waves x 32 queries per workgroup)
  {
    dim3 g(SEQ / 128, HEADS, BATCH);
    attn<<<g, 128, 0, stream>>>(Qh, Kh, Vth, Oh);
  }

  // 4) output projection -> f32
  {
    dim3 g(ROWS / 64, D_MODEL / 128, 1);
    proj_out<<<g, 128, 0, stream>>>(Oh, Woh, bo, out);
  }
}
